// CausalSelfAttention_36747740184916
// MI455X (gfx1250) — matrix operand order repaired and checked by reading, not compiled, not run on hardware
//
#include <hip/hip_runtime.h>
#include <stdint.h>

// ---------------- types ----------------
typedef __attribute__((ext_vector_type(16))) __bf16 v16bf;
typedef __attribute__((ext_vector_type(8)))  float  v8f;
typedef __attribute__((ext_vector_type(4)))  unsigned v4u;
typedef __attribute__((ext_vector_type(4)))  int      v4i;

union Frag { v16bf v; v4u q[2]; };

#define D_MODEL 1024
#define SEQ_T   2048
#define BATCH   4
#define HEADS   16
#define HEAD_D  64
#define M_ROWS  (BATCH * SEQ_T)   // 8192

// ---------------- CDNA5 async global->LDS path (guarded) ----------------
#if defined(__has_builtin)
#if __has_builtin(__builtin_amdgcn_global_load_async_to_lds_b128)
#define HAVE_ASYNC 1
#endif
#endif

typedef __attribute__((address_space(1))) v4i* glb_v4i;
typedef __attribute__((address_space(3))) v4i* lds_v4i;

static __device__ __forceinline__ void async_copy_b128(const uint16_t* g, uint16_t* l) {
#ifdef HAVE_ASYNC
    __builtin_amdgcn_global_load_async_to_lds_b128(
        (glb_v4i)(uintptr_t)g,
        (lds_v4i)(uint32_t)(uintptr_t)l,
        0, 0);
#else
    (void)g; (void)l;
#endif
}

static __device__ __forceinline__ void wait_async0() {
#ifdef HAVE_ASYNC
#if __has_builtin(__builtin_amdgcn_s_wait_asynccnt)
    __builtin_amdgcn_s_wait_asynccnt(0);
#else
    asm volatile("s_wait_asynccnt 0x0" ::: "memory");
#endif
#endif
}

// ---------------- helpers ----------------
static __device__ __forceinline__ uint16_t f2bf(float f) {
    unsigned u = __builtin_bit_cast(unsigned, f);
    unsigned r = (u + 0x7FFFu + ((u >> 16) & 1u)) >> 16;
    return (uint16_t)r;
}

static __device__ __forceinline__ float rmax16(float v) {
#pragma unroll
    for (int m = 1; m < 16; m <<= 1) v = fmaxf(v, __shfl_xor(v, m, 32));
    return v;
}
static __device__ __forceinline__ float rsum16(float v) {
#pragma unroll
    for (int m = 1; m < 16; m <<= 1) v += __shfl_xor(v, m, 32);
    return v;
}

static __device__ __forceinline__ v8f wmma_bf16(const Frag& a, const Frag& b, v8f c) {
    return __builtin_amdgcn_wmma_f32_16x16x32_bf16(false, a.v, false, b.v,
                                                   (short)0, c, false, false);
}

// ---------------- kernel 1: f32 -> bf16 convert ----------------
__global__ __launch_bounds__(256) void cvt_f32_bf16(const float* __restrict__ src,
                                                    uint16_t* __restrict__ dst, int n) {
    int i = blockIdx.x * blockDim.x + threadIdx.x;
    int stride = gridDim.x * blockDim.x;
    for (; i < n; i += stride) dst[i] = f2bf(src[i]);
}

// ---------------- kernels 2/4: tiled bf16 WMMA GEMM ----------------
// C[M x N] = A[M x 1024] * Bw[1024 x N] + bias, block tile 128x128, K-step 64,
// double-buffered LDS, async global->LDS copies, pipeline unrolled x2.
// MODE 0: scatter to Q [B,H,T,D], K-transposed [B,H,D,T], V [B,H,T,D] (bf16).
// MODE 1: f32 output to Co [M x 1024].
template <int MODE>
__global__ __launch_bounds__(256) void gemm_bf16(const uint16_t* __restrict__ A,
                                                 const uint16_t* __restrict__ Bw,
                                                 const float* __restrict__ bias, int N,
                                                 uint16_t* __restrict__ Qo,
                                                 uint16_t* __restrict__ Ko,
                                                 uint16_t* __restrict__ Vo,
                                                 float* __restrict__ Co) {
    __shared__ uint16_t As[2][128 * 64];   // [buf][row][64 k]
    __shared__ uint16_t Bs[2][64 * 128];   // [buf][k][128 n]

    const int t = threadIdx.x;
    const int lane = t & 31, wave = t >> 5;
    const int h = lane >> 4, m = lane & 15;
    const int wm = (wave >> 2) * 64;   // wave row base (2 wave rows)
    const int wn = (wave & 3) * 32;    // wave col base (4 wave cols)
    const int m0 = blockIdx.y * 128, n0 = blockIdx.x * 128;

    v8f acc[4][2];
#pragma unroll
    for (int i = 0; i < 4; ++i)
#pragma unroll
        for (int j = 0; j < 2; ++j) acc[i][j] = v8f{};

    // staging map: A tile 128x64 (64 B/thread), B tile 64x128 (64 B/thread)
    const int ar = t >> 1, ac = (t & 1) * 32;
    const int br = t >> 2, bc = (t & 3) * 32;
    const uint16_t* Ag = A + (size_t)(m0 + ar) * 1024 + ac;
    const uint16_t* Bg = Bw + (size_t)br * N + n0 + bc;

    auto issue_async = [&](int i, int buf) {
        const uint16_t* ga = Ag + i * 64;
        const uint16_t* gb = Bg + (size_t)i * 64 * N;
        uint16_t* la = &As[buf][ar * 64 + ac];
        uint16_t* lb = &Bs[buf][br * 128 + bc];
#pragma unroll
        for (int c = 0; c < 4; ++c) {
            async_copy_b128(ga + c * 8, la + c * 8);
            async_copy_b128(gb + c * 8, lb + c * 8);
        }
    };
    auto stage_sync = [&](int i, int buf) {
        const v4u* ga = reinterpret_cast<const v4u*>(Ag + i * 64);
        const v4u* gb = reinterpret_cast<const v4u*>(Bg + (size_t)i * 64 * N);
        v4u ra[4], rb[4];
#pragma unroll
        for (int c = 0; c < 4; ++c) { ra[c] = ga[c]; rb[c] = gb[c]; }
        v4u* la = reinterpret_cast<v4u*>(&As[buf][ar * 64 + ac]);
        v4u* lb = reinterpret_cast<v4u*>(&Bs[buf][br * 128 + bc]);
#pragma unroll
        for (int c = 0; c < 4; ++c) { la[c] = ra[c]; lb[c] = rb[c]; }
    };
    auto compute = [&](int buf) {
#pragma unroll
        for (int kc = 0; kc < 2; ++kc) {
            Frag bf[2];   // B-fragments hoisted: lane = column N=lane%16, elements over K (table order)
#pragma unroll
            for (int in = 0; in < 2; ++in) {
#pragma unroll
                for (int s = 0; s < 16; ++s)
                    bf[in].v[s] = __builtin_bit_cast(__bf16, Bs[buf][(kc * 32 + 8 * h + s + (s & 8)) * 128 + wn + in * 16 + m]);
            }
#pragma unroll
            for (int im = 0; im < 4; ++im) {
                Frag a;
                const v4u* ap =
                    reinterpret_cast<const v4u*>(&As[buf][(wm + im * 16 + m) * 64 + kc * 32 + 8 * h]);
                a.q[0] = ap[0];   // K = 8h..8h+7
                a.q[1] = ap[2];   // K = 16+8h..
                acc[im][0] = wmma_bf16(a, bf[0], acc[im][0]);
                acc[im][1] = wmma_bf16(a, bf[1], acc[im][1]);
            }
        }
    };

#ifdef HAVE_ASYNC
    issue_async(0, 0);
    // pipeline unrolled x2: constant buffer indices in each half
    for (int i = 0; i < 16; i += 2) {
        wait_async0();
        __syncthreads();
        if (i + 1 < 16) issue_async(i + 1, 1);
        compute(0);
        wait_async0();
        __syncthreads();
        if (i + 2 < 16) issue_async(i + 2, 0);
        compute(1);
    }
#else
    for (int i = 0; i < 16; ++i) {
        __syncthreads();
        stage_sync(i, 0);
        __syncthreads();
        compute(0);
    }
#endif

    // epilogue: C/D layout row = r + 8h, col = m. Branch-free scatter.
#pragma unroll
    for (int im = 0; im < 4; ++im) {
#pragma unroll
        for (int in = 0; in < 2; ++in) {
            const int j = n0 + wn + in * 16 + m;
            const float bj = bias[j];
            if (MODE == 0) {
                const int s = j >> 10;            // 0=q 1=k 2=v
                const int hh = (j >> 6) & 15;     // head
                const int d = j & 63;
                uint16_t* base = (s == 0) ? Qo : ((s == 1) ? Ko : Vo);
#pragma unroll
                for (int r = 0; r < 8; ++r) {
                    const int grow = m0 + wm + im * 16 + r + 8 * h;
                    const int bb = grow >> 11, tq = grow & 2047;
                    const int bhh = bb * HEADS + hh;
                    const size_t idxQV = ((size_t)(bhh * SEQ_T + tq)) * HEAD_D + d;
                    const size_t idxK  = ((size_t)(bhh * HEAD_D + d)) * SEQ_T + tq;
                    const size_t idx = (s == 1) ? idxK : idxQV;
                    base[idx] = f2bf(acc[im][in][r] + bj);
                }
            } else {
#pragma unroll
                for (int r = 0; r < 8; ++r) {
                    const int grow = m0 + wm + im * 16 + r + 8 * h;
                    Co[(size_t)grow * 1024 + j] = acc[im][in][r] + bj;
                }
            }
        }
    }
}

// ---------------- kernel 3: causal flash attention ----------------
// grid: (T/128, B*H). 8 waves; wave w owns query rows q0 + 16w .. +15.
// K pre-transposed in global ([B,H,D,T]); K/V staging double-buffered + async.
__global__ __launch_bounds__(256) void attn_flash(const uint16_t* __restrict__ Q,
                                                  const uint16_t* __restrict__ K,
                                                  const uint16_t* __restrict__ V,
                                                  uint16_t* __restrict__ Y) {
    __shared__ uint16_t Qs[128 * 64];          // query tile [128][64]
    __shared__ uint16_t Kt[2][64 * 32];        // key block transposed [64 d][32 keys]
    __shared__ uint16_t Vs[2][32 * 64];        // value block [32 keys][64 d]
    __shared__ uint16_t Ps[8 * 16 * 32];       // per-wave P tile [16][32]

    const int t = threadIdx.x, lane = t & 31, w = t >> 5;
    const int h = lane >> 4, m = lane & 15;
    const int bh = blockIdx.y, b = bh >> 4, hh = bh & 15;
    const int q0 = blockIdx.x * 128;

    const uint16_t* Qb = Q + (size_t)bh * SEQ_T * HEAD_D;   // [T][64]
    const uint16_t* Kb = K + (size_t)bh * HEAD_D * SEQ_T;   // [64][T]
    const uint16_t* Vb = V + (size_t)bh * SEQ_T * HEAD_D;   // [T][64]

    // staging map (16 B / thread each for K^T and V)
    const int dK = t >> 2, sK = (t & 3) * 8;   // Kt: 64 rows x 32 keys
    const int kV = t >> 3, dV = (t & 7) * 8;   // Vs: 32 keys x 64 d

    auto issue_kv = [&](int kb, int buf) {
        const int kbase = kb * 32;
        const uint16_t* ksrc = Kb + (size_t)dK * SEQ_T + kbase + sK;
        const uint16_t* vsrc = Vb + (size_t)(kbase + kV) * HEAD_D + dV;
#ifdef HAVE_ASYNC
        async_copy_b128(ksrc, &Kt[buf][dK * 32 + sK]);
        async_copy_b128(vsrc, &Vs[buf][kV * 64 + dV]);
#else
        v4u rk = *reinterpret_cast<const v4u*>(ksrc);
        v4u rv = *reinterpret_cast<const v4u*>(vsrc);
        *reinterpret_cast<v4u*>(&Kt[buf][dK * 32 + sK]) = rk;
        *reinterpret_cast<v4u*>(&Vs[buf][kV * 64 + dV]) = rv;
#endif
    };

    {   // load Q tile: thread loads 32 bf16 (64 B)
        const int row = t >> 1, seg = (t & 1) * 32;
        const v4u* src = reinterpret_cast<const v4u*>(Qb + (size_t)(q0 + row) * HEAD_D + seg);
        v4u* dst = reinterpret_cast<v4u*>(&Qs[row * 64 + seg]);
        dst[0] = src[0]; dst[1] = src[1]; dst[2] = src[2]; dst[3] = src[3];
    }

    v8f o[4];
#pragma unroll
    for (int i = 0; i < 4; ++i) o[i] = v8f{};
    float mrow[8], lrow[8];
#pragma unroll
    for (int r = 0; r < 8; ++r) { mrow[r] = -1e30f; lrow[r] = 0.0f; }

    const int kblocks = (q0 + 128) >> 5;       // causal: keys <= q0+127
    const int qmaxw = q0 + w * 16 + 15;        // wave's last query row

    issue_kv(0, 0);
    for (int kb = 0; kb < kblocks; ++kb) {
        const int kbase = kb * 32;
        const int buf = kb & 1;
        wait_async0();            // this wave's block-kb copies landed
        __syncthreads();          // all waves' did; prior compute done
        if (kb + 1 < kblocks) issue_kv(kb + 1, buf ^ 1);
        if (kb + 2 < kblocks) {   // far prefetch hint
            __builtin_prefetch(Kb + (size_t)dK * SEQ_T + kbase + 64 + sK);
            __builtin_prefetch(Vb + (size_t)(kbase + 64 + kV) * HEAD_D + dV);
        }

        if (kbase > qmaxw) continue;   // wave-uniform: whole block masked

        // ---- S = Q K^T (two 16x16 key tiles, D=64 in two K-steps of 32) ----
        v8f c0 = v8f{}, c1 = v8f{};
#pragma unroll
        for (int dblk = 0; dblk < 2; ++dblk) {
            Frag a;
            const v4u* ap = reinterpret_cast<const v4u*>(&Qs[(w * 16 + m) * 64 + dblk * 32 + 8 * h]);
            a.q[0] = ap[0]; a.q[1] = ap[2];
            Frag b0, b1;   // lane = key column lane%16, elements over d (table order)
#pragma unroll
            for (int s = 0; s < 16; ++s) {
                const int kr = (dblk * 32 + 8 * h + s + (s & 8)) * 32;
                b0.v[s] = __builtin_bit_cast(__bf16, Kt[buf][kr + m]);
                b1.v[s] = __builtin_bit_cast(__bf16, Kt[buf][kr + 16 + m]);
            }
            c0 = wmma_bf16(a, b0, c0);
            c1 = wmma_bf16(a, b1, c1);
        }

        // ---- online softmax (rows r+8h, cols = lane%16 per 16-key tile) ----
#pragma unroll
        for (int r = 0; r < 8; ++r) {
            const int q = q0 + w * 16 + r + 8 * h;
            float s0 = c0[r] * 0.125f;                       // 1/sqrt(64)
            if (kbase + m > q) s0 = -1e30f;
            float s1 = c1[r] * 0.125f;
            if (kbase + 16 + m > q) s1 = -1e30f;
            const float mn = fmaxf(mrow[r], rmax16(fmaxf(s0, s1)));
            const float alpha = __expf(mrow[r] - mn);
            const float p0 = __expf(s0 - mn), p1 = __expf(s1 - mn);
            lrow[r] = lrow[r] * alpha + rsum16(p0 + p1);
            mrow[r] = mn;
#pragma unroll
            for (int dt = 0; dt < 4; ++dt) o[dt][r] *= alpha;
            Ps[w * 512 + (r + 8 * h) * 32 + m] = f2bf(p0);
            Ps[w * 512 + (r + 8 * h) * 32 + 16 + m] = f2bf(p1);
        }

        // ---- O += P V ----
        {
            Frag a;
            const v4u* ap = reinterpret_cast<const v4u*>(&Ps[w * 512 + m * 32 + 8 * h]);
            a.q[0] = ap[0]; a.q[1] = ap[2];
#pragma unroll
            for (int dt = 0; dt < 4; ++dt) {
                Frag bv;   // lane = d column lane%16, elements over keys (table order)
#pragma unroll
                for (int s = 0; s < 16; ++s)
                    bv.v[s] = __builtin_bit_cast(__bf16, Vs[buf][(8 * h + s + (s & 8)) * 64 + dt * 16 + m]);
                o[dt] = wmma_bf16(a, bv, o[dt]);
            }
        }
    }

    // ---- normalize and write Y[b*T+q][hh*64 + d] as bf16 ----
#pragma unroll
    for (int r = 0; r < 8; ++r) {
        const int q = q0 + w * 16 + r + 8 * h;
        const float inv = 1.0f / lrow[r];
        const size_t rowoff = ((size_t)(b * SEQ_T + q)) * D_MODEL + hh * HEAD_D;
#pragma unroll
        for (int dt = 0; dt < 4; ++dt)
            Y[rowoff + dt * 16 + m] = f2bf(o[dt][r] * inv);
    }
}

// ---------------- launch ----------------
extern "C" void kernel_launch(void* const* d_in, const int* in_sizes, int n_in,
                              void* d_out, int out_size, void* d_ws, size_t ws_size,
                              hipStream_t stream) {
    (void)in_sizes; (void)n_in; (void)out_size; (void)ws_size;
    const float* x     = (const float*)d_in[0];   // [4,2048,1024]
    const float* wqkv  = (const float*)d_in[1];   // [1024,3072]
    const float* bqkv  = (const float*)d_in[2];   // [3072]
    const float* wproj = (const float*)d_in[3];   // [1024,1024]
    const float* bproj = (const float*)d_in[4];   // [1024]
    float* out = (float*)d_out;                   // [4,2048,1024]

    const size_t MB = 1024ull * 1024ull;
    uint8_t* ws = (uint8_t*)d_ws;
    uint16_t* xb  = (uint16_t*)(ws + 0);          // 16 MB
    uint16_t* wqb = (uint16_t*)(ws + 16 * MB);    //  6 MB
    uint16_t* wpb = (uint16_t*)(ws + 22 * MB);    //  2 MB
    uint16_t* Qo  = (uint16_t*)(ws + 24 * MB);    // 16 MB  [B,H,T,D]
    uint16_t* Ko  = (uint16_t*)(ws + 40 * MB);    // 16 MB  [B,H,D,T] (transposed)
    uint16_t* Vo  = (uint16_t*)(ws + 56 * MB);    // 16 MB  [B,H,T,D]
    uint16_t* Yb  = (uint16_t*)(ws + 72 * MB);    // 16 MB  [M,1024]

    cvt_f32_bf16<<<2048, 256, 0, stream>>>(x,     xb,  M_ROWS * D_MODEL);
    cvt_f32_bf16<<<2048, 256, 0, stream>>>(wqkv,  wqb, D_MODEL * 3 * D_MODEL);
    cvt_f32_bf16<<<1024, 256, 0, stream>>>(wproj, wpb, D_MODEL * D_MODEL);

    // QKV GEMM: M=8192, N=3072
    gemm_bf16<0><<<dim3(3 * D_MODEL / 128, M_ROWS / 128), 256, 0, stream>>>(
        xb, wqb, bqkv, 3 * D_MODEL, Qo, Ko, Vo, nullptr);

    // flash attention: (T/128) x (B*H)
    attn_flash<<<dim3(SEQ_T / 128, BATCH * HEADS), 256, 0, stream>>>(Qo, Ko, Vo, Yb);

    // projection GEMM: M=8192, N=1024, f32 out
    gemm_bf16<1><<<dim3(D_MODEL / 128, M_ROWS / 128), 256, 0, stream>>>(
        Yb, wpb, bproj, D_MODEL, nullptr, nullptr, nullptr, out);
}
